// EntityAttentionLayer_2551210573963
// MI455X (gfx1250) — compile-verified, matrix operand order checked by reading, not run on hardware
//
#include <hip/hip_runtime.h>
#include <hip/hip_bf16.h>
#include <stdint.h>

// ---------------------------------------------------------------------------
// EntityAttention for MI455X (gfx1250): bf16 WMMA everywhere, fp32 softmax.
//   BS=128, NE=512, NQ=128, IN_DIM=512, EMBED=512, H=8, HD=64
// ---------------------------------------------------------------------------

typedef __bf16 bf16_t;
typedef __attribute__((ext_vector_type(16))) __bf16 v16bf;
typedef __attribute__((ext_vector_type(8)))  float  v8f;

#define BS     128
#define NE     512
#define NQJ    128
#define IN_DIM 512
#define EMBED  512
#define HEADS  8
#define HD     64

#define LDT 40    // bf16 LDS row stride for 32-wide K tiles (80B: 16B-aligned rows)
#define LDW 520   // bf16 LDS row stride for 512-wide weight rows (1040B, 16B-aligned)

__device__ __forceinline__ bf16_t f2bf(float f) {
  uint32_t u = __builtin_bit_cast(uint32_t, f);
  uint32_t r = u + 0x7FFFu + ((u >> 16) & 1u);   // round-to-nearest-even
  uint16_t h = (uint16_t)(r >> 16);
  return __builtin_bit_cast(bf16_t, h);
}
__device__ __forceinline__ float bf2f(bf16_t x) {
  uint16_t s = __builtin_bit_cast(uint16_t, x);
  uint32_t u = ((uint32_t)s) << 16;
  return __builtin_bit_cast(float, u);
}

union FragU { v16bf v; uint4 q[2]; };

// Load one 16x32 bf16 WMMA operand fragment.  Per the CDNA5 16-bit A/B layout:
// lane L<16 holds row L chunks K[0..7] and K[16..23]; lane 16+L holds row L
// chunks K[8..15] and K[24..31].  Caller passes p already offset by
// row*stride + (lane>>4)*8; we grab 16B at p and 16B at p+16 elements.
__device__ __forceinline__ v16bf load16(const bf16_t* p) {
  FragU u;
  u.q[0] = *reinterpret_cast<const uint4*>(p);
  u.q[1] = *reinterpret_cast<const uint4*>(p + 16);
  return u.v;
}

__device__ __forceinline__ v8f wmma_bf16(v16bf a, v16bf b, v8f c) {
  return __builtin_amdgcn_wmma_f32_16x16x32_bf16(
      /*neg_a=*/false, a, /*neg_b=*/false, b,
      /*c_mod=*/(short)0, c, /*reuse_a=*/false, /*reuse_b=*/false);
}

// ---------------------------------------------------------------------------
// Kernel 0: fp32 -> bf16 conversion (entities + weights; grid-stride)
// ---------------------------------------------------------------------------
__global__ __launch_bounds__(256)
void cvt_kernel(const float* __restrict__ src, bf16_t* __restrict__ dst, int n) {
  int i = blockIdx.x * blockDim.x + threadIdx.x;
  int stride = gridDim.x * blockDim.x;
  for (; i < n; i += stride) dst[i] = f2bf(src[i]);
}

// ---------------------------------------------------------------------------
// Kernel 1: QKV = E(65536x512) * W_in^T(512x1536), routed into Q/K/Vt buffers.
//   grid (12, 512); 128x128 tile per block, 8 waves, K in 32-steps via LDS.
//   Q tiles with entity index >= NQ are skipped (3/4 of Q compute).
// ---------------------------------------------------------------------------
__global__ __launch_bounds__(256)
void qkv_gemm_kernel(const bf16_t* __restrict__ E, const bf16_t* __restrict__ W,
                     bf16_t* __restrict__ Q, bf16_t* __restrict__ K,
                     bf16_t* __restrict__ Vt)
{
  __shared__ bf16_t As[128 * LDT];
  __shared__ bf16_t Bsh[128 * LDT];
  const int x = blockIdx.x, y = blockIdx.y;
  if (x < 4 && (y & 3) != 0) return;     // Q region, rows with n>=NQ: dead output
  const int m0 = y * 128, n0 = x * 128;
  const int tid  = threadIdx.x;
  const int lane = tid & 31, wv = tid >> 5;
  const int colL = lane & 15, half = lane >> 4, hs = half * 8;
  const int lr = tid >> 1, lc = (tid & 1) * 16;   // cooperative tile loader

  v8f acc[8] = {};

  for (int kt = 0; kt < 16; ++kt) {
    const int k0 = kt * 32;
    __syncthreads();
    {
      const uint4* ga = reinterpret_cast<const uint4*>(E + (size_t)(m0 + lr) * IN_DIM + k0 + lc);
      const uint4* gb = reinterpret_cast<const uint4*>(W + (size_t)(n0 + lr) * IN_DIM + k0 + lc);
      uint4 a0 = ga[0], a1 = ga[1];
      uint4 b0 = gb[0], b1 = gb[1];
      if (kt < 15) {   // prefetch next K tile (global_prefetch_b8)
        __builtin_prefetch(E + (size_t)(m0 + lr) * IN_DIM + k0 + 32 + lc, 0, 1);
        __builtin_prefetch(W + (size_t)(n0 + lr) * IN_DIM + k0 + 32 + lc, 0, 1);
      }
      uint4* la = reinterpret_cast<uint4*>(As + lr * LDT + lc);
      uint4* lb = reinterpret_cast<uint4*>(Bsh + lr * LDT + lc);
      la[0] = a0; la[1] = a1;
      lb[0] = b0; lb[1] = b1;
    }
    __syncthreads();
    v16bf a = load16(As + (wv * 16 + colL) * LDT + hs);
    #pragma unroll
    for (int j = 0; j < 8; ++j) {
      v16bf bfr = load16(Bsh + (j * 16 + colL) * LDT + hs);
      acc[j] = wmma_bf16(a, bfr, acc[j]);
    }
  }

  // Epilogue: C/D layout -> VGPR v holds row (v + 8*half), lane column colL.
  #pragma unroll
  for (int j = 0; j < 8; ++j) {
    const int f = n0 + j * 16 + colL;
    #pragma unroll
    for (int v = 0; v < 8; ++v) {
      const int m = m0 + wv * 16 + half * 8 + v;
      const int b = m >> 9, n = m & 511;
      const float val = acc[j][v];
      if (f < EMBED) {                       // Q (scaled by 1/sqrt(HD))
        const int h = f >> 6, d = f & 63;
        Q[(((size_t)(b * HEADS + h)) * NQJ + n) * HD + d] = f2bf(val * 0.125f);
      } else if (f < 2 * EMBED) {            // K: [b][h][ne][hd]
        const int fk = f - EMBED, h = fk >> 6, d = fk & 63;
        K[(((size_t)(b * HEADS + h)) * NE + n) * HD + d] = f2bf(val);
      } else {                               // V transposed: [b][h][hd][ne]
        const int fv = f - 2 * EMBED, h = fv >> 6, d = fv & 63;
        Vt[(((size_t)(b * HEADS + h)) * HD + d) * NE + n] = f2bf(val);
      }
    }
  }
}

// ---------------------------------------------------------------------------
// Kernel 2: per (b,h) fused attention.  1024 blocks, 8 waves each.
//   phase 1: logits(128x512) = Q*K^T -> LDS (bf16)
//   phase 2: pre-mask, softmax (NaN->0), diff-mask renorm (fp32 in regs)
//   phase 3: attn(128x64) = w * V  (A-fragments straight from LDS)
// ---------------------------------------------------------------------------
__global__ __launch_bounds__(256)
void attn_kernel(const bf16_t* __restrict__ Q, const bf16_t* __restrict__ K,
                 const bf16_t* __restrict__ Vt, const int* __restrict__ pre_mask,
                 const float* __restrict__ diff_mask, bf16_t* __restrict__ Aout)
{
  extern __shared__ bf16_t lw[];   // [128][LDW] bf16 logits/weights (~130 KB)
  const int bh = blockIdx.x;
  const int b = bh >> 3, h = bh & 7;
  const int tid = threadIdx.x, lane = tid & 31, w = tid >> 5;
  const int colL = lane & 15, half = lane >> 4, hs = half * 8;

  const bf16_t* Qb = Q  + (size_t)bh * NQJ * HD;
  const bf16_t* Kb = K  + (size_t)bh * NE  * HD;
  const bf16_t* Vb = Vt + (size_t)bh * HD  * NE;

  // ---- phase 1: logits.  Wave w owns columns [w*64, w*64+64).
  v16bf kf[4][2];
  #pragma unroll
  for (int j = 0; j < 4; ++j)
    #pragma unroll
    for (int ks = 0; ks < 2; ++ks) {
      const int ne = w * 64 + j * 16 + colL;
      kf[j][ks] = load16(Kb + (size_t)ne * HD + ks * 32 + hs);
    }
  for (int mt = 0; mt < 8; ++mt) {
    v8f acc[4] = {};
    #pragma unroll
    for (int ks = 0; ks < 2; ++ks) {
      v16bf a = load16(Qb + (size_t)(mt * 16 + colL) * HD + ks * 32 + hs);
      #pragma unroll
      for (int j = 0; j < 4; ++j) acc[j] = wmma_bf16(a, kf[j][ks], acc[j]);
    }
    #pragma unroll
    for (int j = 0; j < 4; ++j)
      #pragma unroll
      for (int v = 0; v < 8; ++v) {
        const int r = mt * 16 + half * 8 + v;
        const int c = w * 64 + j * 16 + colL;
        lw[r * LDW + c] = f2bf(acc[j][v]);
      }
  }
  __syncthreads();

  // ---- phase 2: softmax.  Wave w owns rows [w*16, w*16+16); lane strides 32.
  const float NEGINF = -__builtin_inff();
  for (int i = 0; i < 16; ++i) {
    const int r = w * 16 + i;
    const int*   pm = pre_mask  + ((size_t)b * NQJ + r) * NE;
    const float* dm = diff_mask + ((size_t)b * NQJ + r) * NE;
    float vals[16];
    float mx = NEGINF;
    #pragma unroll
    for (int jj = 0; jj < 16; ++jj) {
      const int c = jj * 32 + lane;
      float xv = bf2f(lw[r * LDW + c]);
      xv = pm[c] ? NEGINF : xv;
      vals[jj] = xv;
      mx = fmaxf(mx, xv);
    }
    #pragma unroll
    for (int off = 16; off > 0; off >>= 1) mx = fmaxf(mx, __shfl_xor(mx, off));
    float s = 0.f;
    #pragma unroll
    for (int jj = 0; jj < 16; ++jj) {
      const float e = (vals[jj] == NEGINF) ? 0.f : __expf(vals[jj] - mx);
      vals[jj] = e; s += e;
    }
    #pragma unroll
    for (int off = 16; off > 0; off >>= 1) s += __shfl_xor(s, off);
    const float inv = (s > 0.f) ? 1.f / s : 0.f;   // fully-masked row -> all 0
    float s2 = 0.f;
    #pragma unroll
    for (int jj = 0; jj < 16; ++jj) {
      const int c = jj * 32 + lane;
      const float wt = vals[jj] * inv * dm[c];
      vals[jj] = wt; s2 += wt;
    }
    #pragma unroll
    for (int off = 16; off > 0; off >>= 1) s2 += __shfl_xor(s2, off);
    const float inv2 = 1.f / (s2 + 1e-8f);
    #pragma unroll
    for (int jj = 0; jj < 16; ++jj) {
      const int c = jj * 32 + lane;
      lw[r * LDW + c] = f2bf(vals[jj] * inv2);
    }
  }
  __syncthreads();

  // ---- phase 3: attn = w(128x512) * V(512x64).  Wave w owns rows [w*16,..).
  v8f acc2[4] = {};
  for (int ks = 0; ks < 16; ++ks) {
    const int k0 = ks * 32;
    v16bf a = load16(lw + (size_t)(w * 16 + colL) * LDW + k0 + hs);
    #pragma unroll
    for (int j = 0; j < 4; ++j) {
      const int d = j * 16 + colL;
      v16bf bb = load16(Vb + (size_t)d * NE + k0 + hs);
      acc2[j] = wmma_bf16(a, bb, acc2[j]);
    }
  }
  #pragma unroll
  for (int j = 0; j < 4; ++j)
    #pragma unroll
    for (int v = 0; v < 8; ++v) {
      const int q = w * 16 + half * 8 + v;
      const int d = j * 16 + colL;
      Aout[((size_t)(b * NQJ + q)) * EMBED + h * HD + d] = f2bf(acc2[j][v]);
    }
}

// ---------------------------------------------------------------------------
// Kernel 3: out(16384x512) = attn * W_out^T + b_out, then post-mask zeroing.
//   grid (4, 128); same LDS-tiled WMMA structure as kernel 1.
// ---------------------------------------------------------------------------
__global__ __launch_bounds__(256)
void out_gemm_kernel(const bf16_t* __restrict__ A, const bf16_t* __restrict__ W,
                     const float* __restrict__ b_out, const int* __restrict__ post_mask,
                     float* __restrict__ out)
{
  __shared__ bf16_t As[128 * LDT];
  __shared__ bf16_t Bsh[128 * LDT];
  const int x = blockIdx.x, y = blockIdx.y;
  const int m0 = y * 128, n0 = x * 128;
  const int tid  = threadIdx.x;
  const int lane = tid & 31, wv = tid >> 5;
  const int colL = lane & 15, half = lane >> 4, hs = half * 8;
  const int lr = tid >> 1, lc = (tid & 1) * 16;

  v8f acc[8] = {};

  for (int kt = 0; kt < 16; ++kt) {
    const int k0 = kt * 32;
    __syncthreads();
    {
      const uint4* ga = reinterpret_cast<const uint4*>(A + (size_t)(m0 + lr) * EMBED + k0 + lc);
      const uint4* gb = reinterpret_cast<const uint4*>(W + (size_t)(n0 + lr) * EMBED + k0 + lc);
      uint4 a0 = ga[0], a1 = ga[1];
      uint4 b0 = gb[0], b1 = gb[1];
      uint4* la = reinterpret_cast<uint4*>(As + lr * LDT + lc);
      uint4* lb = reinterpret_cast<uint4*>(Bsh + lr * LDT + lc);
      la[0] = a0; la[1] = a1;
      lb[0] = b0; lb[1] = b1;
    }
    __syncthreads();
    v16bf a = load16(As + (wv * 16 + colL) * LDT + hs);
    #pragma unroll
    for (int j = 0; j < 8; ++j) {
      v16bf bfr = load16(Bsh + (j * 16 + colL) * LDT + hs);
      acc[j] = wmma_bf16(a, bfr, acc[j]);
    }
  }

  #pragma unroll
  for (int j = 0; j < 8; ++j) {
    const int f = n0 + j * 16 + colL;
    const float bo = b_out[f];
    #pragma unroll
    for (int v = 0; v < 8; ++v) {
      const int m = m0 + wv * 16 + half * 8 + v;
      const int bb = m >> 7, qi = m & 127;
      float val = acc[j][v] + bo;
      if (post_mask[bb * NQJ + qi] != 0) val = 0.f;
      out[(size_t)m * EMBED + f] = val;
    }
  }
}

// ---------------------------------------------------------------------------
extern "C" void kernel_launch(void* const* d_in, const int* in_sizes, int n_in,
                              void* d_out, int out_size, void* d_ws, size_t ws_size,
                              hipStream_t stream)
{
  (void)in_sizes; (void)n_in; (void)out_size; (void)ws_size;
  const float* entities  = (const float*)d_in[0];
  const int*   pre_mask  = (const int*)  d_in[1];
  const float* diff_mask = (const float*)d_in[2];
  const int*   post_mask = (const int*)  d_in[3];
  const float* W_in      = (const float*)d_in[4];
  const float* W_out     = (const float*)d_in[5];
  const float* b_out     = (const float*)d_in[6];
  float* out = (float*)d_out;

  char* p = (char*)d_ws;
  auto alloc = [&](size_t bytes) {
    char* r = p;
    p += (bytes + 255) & ~(size_t)255;
    return r;
  };
  bf16_t* Ebf    = (bf16_t*)alloc((size_t)BS * NE * IN_DIM * 2);
  bf16_t* Winbf  = (bf16_t*)alloc((size_t)3 * EMBED * IN_DIM * 2);
  bf16_t* Woutbf = (bf16_t*)alloc((size_t)EMBED * EMBED * 2);
  bf16_t* Qbf    = (bf16_t*)alloc((size_t)BS * HEADS * NQJ * HD * 2);
  bf16_t* Kbf    = (bf16_t*)alloc((size_t)BS * HEADS * NE * HD * 2);
  bf16_t* Vtbf   = (bf16_t*)alloc((size_t)BS * HEADS * HD * NE * 2);
  bf16_t* Abuf   = (bf16_t*)alloc((size_t)BS * NQJ * EMBED * 2);

  (void)hipFuncSetAttribute(reinterpret_cast<const void*>(attn_kernel),
                            hipFuncAttributeMaxDynamicSharedMemorySize,
                            128 * LDW * 2);

  cvt_kernel<<<2048, 256, 0, stream>>>(entities, Ebf, BS * NE * IN_DIM);
  cvt_kernel<<<512,  256, 0, stream>>>(W_in,  Winbf,  3 * EMBED * IN_DIM);
  cvt_kernel<<<256,  256, 0, stream>>>(W_out, Woutbf, EMBED * EMBED);

  qkv_gemm_kernel<<<dim3(12, 512), 256, 0, stream>>>(Ebf, Winbf, Qbf, Kbf, Vtbf);

  attn_kernel<<<dim3(BS * HEADS), 256, 128 * LDW * 2, stream>>>(
      Qbf, Kbf, Vtbf, pre_mask, diff_mask, Abuf);

  out_gemm_kernel<<<dim3(4, 128), 256, 0, stream>>>(
      Abuf, Woutbf, b_out, post_mask, out);
}
